// EfficientSpatialWindowAttention_9749575762710
// MI455X (gfx1250) — compile-verified, matrix operand order checked by reading, not run on hardware
//
#include <hip/hip_runtime.h>
#include <hip/hip_bf16.h>

typedef __bf16 bf16_t;
typedef __attribute__((ext_vector_type(16))) __bf16 v16bf;
typedef __attribute__((ext_vector_type(8)))  __bf16 v8bf;
typedef __attribute__((ext_vector_type(8)))  float  v8f;
typedef __attribute__((ext_vector_type(4)))  float  v4f;

#define NH      12
#define DH      64
#define NTOK    49        // tokens per window
#define TPAD    64        // padded token axis
#define LK      64        // Linformer compressed length
#define BANKN   16
#define KVLEN   80        // LK + BANKN
#define KVPAD   96
#define CDIM    768
#define C3      2304
#define MTOT    50176     // BW * 49
#define BWH     12288     // BW * NH

// ---------------- helpers ----------------

__device__ __forceinline__ v8f vzero8() {
  v8f z;
#pragma unroll
  for (int i = 0; i < 8; ++i) z[i] = 0.0f;
  return z;
}

__device__ __forceinline__ v8bf bzero8() {
  v8bf z;
#pragma unroll
  for (int i = 0; i < 8; ++i) z[i] = (bf16_t)0.0f;
  return z;
}

__device__ __forceinline__ v8bf cvt8(const float* __restrict__ p) {
  v8bf t;
#pragma unroll
  for (int i = 0; i < 8; ++i) t[i] = (bf16_t)p[i];
  return t;
}

__device__ __forceinline__ v8bf cvt44(v4f lo, v4f hi) {
  v8bf t;
#pragma unroll
  for (int i = 0; i < 4; ++i) {
    t[i]     = (bf16_t)lo[i];
    t[4 + i] = (bf16_t)hi[i];
  }
  return t;
}

__device__ __forceinline__ v8f wmma_bf16(v16bf a, v16bf b, v8f c) {
  return __builtin_amdgcn_wmma_f32_16x16x32_bf16(false, a, false, b, (short)0, c,
                                                 false, false);
}

// Fragment load, AMD table K order (k = 8h+s, then 16+8h+s-8; two 16-byte loads).
__device__ __forceinline__ v16bf frag_ld(const bf16_t* rowbase, int kbase, int lane) {
  const bf16_t* p = rowbase + kbase + ((lane & 16) ? 8 : 0);
  v8bf lo = *(const v8bf*)(p);
  v8bf hi = *(const v8bf*)(p + 16);
  return __builtin_shufflevector(lo, hi, 0, 1, 2, 3, 4, 5, 6, 7,
                                 8, 9, 10, 11, 12, 13, 14, 15);
}

// ---- CDNA5 async global->LDS copy path (ASYNCcnt-tracked, bypasses VGPRs) ----
// Flat shared-aperture address: aperture match is on ADDR[63:32]; low 32 bits
// carry the LDS byte offset, which is what the instruction's VDST VGPR takes.
__device__ __forceinline__ unsigned lds_off(const void* p) {
  return (unsigned)(unsigned long long)p;
}

__device__ __forceinline__ void async_b128(unsigned lds, const void* g) {
  asm volatile("global_load_async_to_lds_b128 %0, %1, off"
               :: "v"(lds), "v"(g) : "memory");
}

__device__ __forceinline__ void wait_async() {
  asm volatile("s_wait_asynccnt 0x0" ::: "memory");
}

// token row (bw*49+t) -> row index into x (b*3136 + n), window partition fused
__device__ __forceinline__ int token_to_xrow(int r) {
  int bw = r / 49, t = r - bw * 49;
  int b  = bw >> 6, wi = bw & 63;
  int wy = wi >> 3, wx = wi & 7;
  int ty = t / 7,  tx = t - ty * 7;
  return b * 3136 + (wy * 7 + ty) * 56 + (wx * 7 + tx);
}

// ---------------- prep: bf16 conversion / transposes ----------------

__global__ __launch_bounds__(256) void prep_kernel(
    const float* __restrict__ qkv_w, const float* __restrict__ proj_w,
    const float* __restrict__ Ek,    const float* __restrict__ Ev,
    const float* __restrict__ gk,    const float* __restrict__ gv,
    bf16_t* wqT, bf16_t* wpT, bf16_t* ekt, bf16_t* evt, bf16_t* gkb, bf16_t* gvb) {
  const long NQ = (long)CDIM * C3;
  const long NP = (long)CDIM * CDIM;
  const long NE = 64 * 64;
  const long NG = NH * BANKN * DH;
  const long TOT = NQ + NP + 2 * NE + 2 * NG;
  long stride = (long)gridDim.x * blockDim.x;
  for (long idx = (long)blockIdx.x * blockDim.x + threadIdx.x; idx < TOT; idx += stride) {
    long t = idx;
    if (t < NQ) {  // wqT [N=2304][K=768] <- qkv_w [768][2304]
      int n = (int)(t / CDIM), k = (int)(t - (long)n * CDIM);
      wqT[t] = (bf16_t)qkv_w[(long)k * C3 + n];
      continue;
    }
    t -= NQ;
    if (t < NP) {  // wpT [768][768] transposed
      int n = (int)(t / CDIM), k = (int)(t - (long)n * CDIM);
      wpT[t] = (bf16_t)proj_w[(long)k * CDIM + n];
      continue;
    }
    t -= NP;
    if (t < NE) {  // E_k^T: [Lk=64][n pad 64]
      int kk = (int)(t >> 6), n = (int)(t & 63);
      ekt[t] = (n < NTOK) ? (bf16_t)Ek[n * LK + kk] : (bf16_t)0.0f;
      continue;
    }
    t -= NE;
    if (t < NE) {
      int kk = (int)(t >> 6), n = (int)(t & 63);
      evt[t] = (n < NTOK) ? (bf16_t)Ev[n * LK + kk] : (bf16_t)0.0f;
      continue;
    }
    t -= NE;
    if (t < NG) {  // global_k [bank][C] -> [h][bank][dh]
      int h = (int)(t >> 10), r = (int)((t >> 6) & 15), d = (int)(t & 63);
      gkb[t] = (bf16_t)gk[r * CDIM + h * DH + d];
      continue;
    }
    t -= NG;
    {
      int h = (int)(t >> 10), r = (int)((t >> 6) & 15), d = (int)(t & 63);
      gvb[t] = (bf16_t)gv[r * CDIM + h * DH + d];
    }
  }
}

// zero the padded token slots (t in [49,64)) of the transposed k/v buffers
__global__ __launch_bounds__(256) void pad_kv_kernel(bf16_t* kT, bf16_t* vT) {
  const long TOT = (long)BWH * DH * (TPAD - NTOK);
  long stride = (long)gridDim.x * blockDim.x;
  for (long idx = (long)blockIdx.x * blockDim.x + threadIdx.x; idx < TOT; idx += stride) {
    long q = idx / (TPAD - NTOK);                 // bwh*64 + d
    int  t = NTOK + (int)(idx - q * (TPAD - NTOK));
    kT[q * TPAD + t] = (bf16_t)0.0f;
    vT[q * TPAD + t] = (bf16_t)0.0f;
  }
}

// ---------------- GEMM 1: qkv = window_partition(x) @ qkv_w + b ----------------
// 128x128x32 tiles, double-buffered LDS; B tiles via async global->LDS,
// A tiles via raw-float register prefetch (convert sunk past the WMMA block).

__global__ __launch_bounds__(256) void gemm_qkv_kernel(
    const float* __restrict__ x, const bf16_t* __restrict__ wqT,
    const float* __restrict__ bias,
    bf16_t* __restrict__ qb, bf16_t* __restrict__ kT, bf16_t* __restrict__ vT) {
  __shared__ alignas(16) bf16_t As[2][128 * 40];
  __shared__ alignas(16) bf16_t Bs[2][128 * 40];
  const int tid = threadIdx.x, lane = tid & 31, wave = tid >> 5;
  const int Mbase = blockIdx.x * 128, Nbase = blockIdx.y * 128;
  const int wm = wave & 3, wn = wave >> 2;

  // per-thread copy slots: rows r and r+64, 8-element chunk at column c
  const int r = tid >> 2, c = (tid & 3) * 8;
  const float*  __restrict__ srcA0 = x + (long)token_to_xrow(Mbase + r) * CDIM + c;
  const float*  __restrict__ srcA1 = x + (long)token_to_xrow(Mbase + r + 64) * CDIM + c;
  const bf16_t* __restrict__ srcB0 = wqT + (long)(Nbase + r) * CDIM + c;
  const bf16_t* __restrict__ srcB1 = wqT + (long)(Nbase + r + 64) * CDIM + c;
  const int sA0 = r * 40 + c, sA1 = (r + 64) * 40 + c;
  const unsigned BUFSZ = 128 * 40 * 2;              // bytes per LDS buffer
  const unsigned ldsB0 = lds_off(&Bs[0][sA0]);
  const unsigned ldsB1 = lds_off(&Bs[0][sA1]);

  // preload tile 0 (B async, A via registers)
  async_b128(ldsB0, srcB0);
  async_b128(ldsB1, srcB1);
  *(v8bf*)&As[0][sA0] = cvt8(srcA0);
  *(v8bf*)&As[0][sA1] = cvt8(srcA1);
  wait_async();
  __syncthreads();

  v8f acc[2][4];
#pragma unroll
  for (int i = 0; i < 2; ++i)
#pragma unroll
    for (int j = 0; j < 4; ++j) acc[i][j] = vzero8();

  const int arow0 = (wm * 32 + (lane & 15)) * 40;
  const int brow0 = (wn * 64 + (lane & 15)) * 40;

  int buf = 0;
  for (int kb0 = 0; kb0 < CDIM - 32; kb0 += 32) {
    const int knx = kb0 + 32;
    const unsigned nb = (unsigned)(buf ^ 1) * BUFSZ;
    async_b128(ldsB0 + nb, srcB0 + knx);            // next B tile, no VGPR traffic
    async_b128(ldsB1 + nb, srcB1 + knx);
    // issue raw A loads now; convert AFTER the WMMA block so they stay in flight
    v4f ra0 = *(const v4f*)(srcA0 + knx);
    v4f ra1 = *(const v4f*)(srcA0 + knx + 4);
    v4f ra2 = *(const v4f*)(srcA1 + knx);
    v4f ra3 = *(const v4f*)(srcA1 + knx + 4);

    const bf16_t* Ab = As[buf];
    const bf16_t* Bb = Bs[buf];
    v16bf af[2], bfr[4];
#pragma unroll
    for (int i = 0; i < 2; ++i) af[i] = frag_ld(Ab + arow0 + i * 16 * 40, 0, lane);
#pragma unroll
    for (int j = 0; j < 4; ++j) bfr[j] = frag_ld(Bb + brow0 + j * 16 * 40, 0, lane);
#pragma unroll
    for (int i = 0; i < 2; ++i)
#pragma unroll
      for (int j = 0; j < 4; ++j) acc[i][j] = wmma_bf16(af[i], bfr[j], acc[i][j]);

    bf16_t* An = As[buf ^ 1];
    *(v8bf*)&An[sA0] = cvt44(ra0, ra1);
    *(v8bf*)&An[sA1] = cvt44(ra2, ra3);
    wait_async();
    __syncthreads();
    buf ^= 1;
  }
  {  // final k-step (no prefetch)
    const bf16_t* Ab = As[buf];
    const bf16_t* Bb = Bs[buf];
    v16bf af[2], bfr[4];
#pragma unroll
    for (int i = 0; i < 2; ++i) af[i] = frag_ld(Ab + arow0 + i * 16 * 40, 0, lane);
#pragma unroll
    for (int j = 0; j < 4; ++j) bfr[j] = frag_ld(Bb + brow0 + j * 16 * 40, 0, lane);
#pragma unroll
    for (int i = 0; i < 2; ++i)
#pragma unroll
      for (int j = 0; j < 4; ++j) acc[i][j] = wmma_bf16(af[i], bfr[j], acc[i][j]);
  }

  // epilogue: one /49 per tile-row, incremental carry for the rest
  const int roff = (lane & 16) ? 8 : 0;
  const int cl = lane & 15;
#pragma unroll
  for (int i = 0; i < 2; ++i) {
    int row0 = Mbase + wm * 32 + i * 16 + roff;
    int bw = row0 / 49, t = row0 - bw * 49;
    int bwv[8], tv[8];
#pragma unroll
    for (int e = 0; e < 8; ++e) {
      bwv[e] = bw; tv[e] = t;
      if (++t == NTOK) { t = 0; ++bw; }
    }
#pragma unroll
    for (int j = 0; j < 4; ++j) {
      int cc = Nbase + wn * 64 + j * 16 + cl;          // column in [0, 2304)
      float bv = bias[cc];
      int which = cc / CDIM, rem = cc - which * CDIM;
      int h = rem >> 6, d = rem & 63;
#pragma unroll
      for (int e = 0; e < 8; ++e) {
        long bwh = (long)(bwv[e] * NH + h);
        bf16_t val = (bf16_t)(acc[i][j][e] + bv);
        if (which == 0)      qb[(bwh * NTOK + tv[e]) * DH + d] = val;
        else if (which == 1) kT[(bwh * DH + d) * TPAD + tv[e]] = val;
        else                 vT[(bwh * DH + d) * TPAD + tv[e]] = val;
      }
    }
  }
}

// ---------------- fused compression + attention, one block per (window, head) ----------------

__global__ __launch_bounds__(128) void attn_kernel(
    const bf16_t* __restrict__ qb, const bf16_t* __restrict__ kTb,
    const bf16_t* __restrict__ vTb,
    const bf16_t* __restrict__ ekt, const bf16_t* __restrict__ evt,
    const bf16_t* __restrict__ gkb, const bf16_t* __restrict__ gvb,
    bf16_t* __restrict__ aout) {
  __shared__ alignas(16) bf16_t qs[64 * 64];       // [m][d]
  __shared__ alignas(16) bf16_t ksT[64 * 64];      // [d][n]
  __shared__ alignas(16) bf16_t vsT[64 * 64];      // [d][n]
  __shared__ alignas(16) bf16_t kfull[KVLEN * 64]; // [key][d]
  __shared__ alignas(16) bf16_t vfullT[64 * KVPAD];// [d][key96]
  __shared__ alignas(16) bf16_t probs[64 * KVPAD]; // [m][key96]

  const int tid = threadIdx.x, lane = tid & 31, wave = tid >> 5;
  const int bwh = blockIdx.x;
  const int bw = bwh / NH, h = bwh - bw * NH;
  const bf16_t* qsrc  = qb  + (long)bwh * NTOK * DH;
  const bf16_t* ksrcT = kTb + (long)bwh * DH * TPAD;
  const bf16_t* vsrcT = vTb + (long)bwh * DH * TPAD;
  const v8bf z8 = bzero8();

  // ---- phase 1: async staging (pure copies bypass VGPRs) ----
  for (int ch = tid; ch < 512; ch += 128) {
    int rr = ch >> 3, ck = (ch & 7) * 8;
    if (rr < NTOK) async_b128(lds_off(&qs[rr * 64 + ck]), qsrc + rr * DH + ck);
    else           *(v8bf*)&qs[rr * 64 + ck] = z8;
    async_b128(lds_off(&ksT[rr * 64 + ck]), ksrcT + rr * TPAD + ck);
    async_b128(lds_off(&vsT[rr * 64 + ck]), vsrcT + rr * TPAD + ck);
  }
  {                                                  // bank rows of kfull (16x64)
    int rr = tid >> 3, ck = (tid & 7) * 8;
    async_b128(lds_off(&kfull[(LK + rr) * 64 + ck]), gkb + (h * BANKN + rr) * DH + ck);
  }
  for (int idx = tid; idx < BANKN * 64; idx += 128) {  // bank cols of vfullT
    int rr = idx >> 6, d = idx & 63;
    vfullT[d * KVPAD + LK + rr] = gvb[(h * BANKN + rr) * DH + d];
  }
  {                                                  // vfullT pad keys 80..95
    int d = tid >> 1, kk = KVLEN + (tid & 1) * 8;
    *(v8bf*)&vfullT[d * KVPAD + kk] = z8;
  }
  for (int ch = tid; ch < 768; ch += 128)            // zero probs (incl. pad cols)
    *(v8bf*)&probs[ch * 8] = z8;
  wait_async();
  __syncthreads();

  const int mb = wave * 16;
  const int roff = (lane & 16) ? 8 : 0;
  const int cl = lane & 15;

  // ---- phase 2: k_c = EkT @ k, v_c = EvT @ v ----
  {
    v16bf ae[2], av[2];
#pragma unroll
    for (int kk = 0; kk < 2; ++kk) {                 // A frags are j-invariant
      ae[kk] = frag_ld(ekt + (mb + cl) * 64, kk * 32, lane);
      av[kk] = frag_ld(evt + (mb + cl) * 64, kk * 32, lane);
    }
#pragma unroll
    for (int j = 0; j < 4; ++j) {
      v8f ak = vzero8(), avv = vzero8();
#pragma unroll
      for (int kk = 0; kk < 2; ++kk) {
        v16bf bk = frag_ld(&ksT[(j * 16 + cl) * 64], kk * 32, lane);
        ak = wmma_bf16(ae[kk], bk, ak);
        v16bf bv = frag_ld(&vsT[(j * 16 + cl) * 64], kk * 32, lane);
        avv = wmma_bf16(av[kk], bv, avv);
      }
#pragma unroll
      for (int e = 0; e < 8; ++e) {
        kfull[(mb + roff + e) * 64 + j * 16 + cl]     = (bf16_t)ak[e];
        vfullT[(j * 16 + cl) * KVPAD + mb + roff + e] = (bf16_t)avv[e];
      }
    }
  }
  __syncthreads();

  // ---- phase 3: scores = q @ k_full^T * scale, softmax over 80 keys ----
  v8f sc[5];
  {
    v16bf aq[2];
#pragma unroll
    for (int kk = 0; kk < 2; ++kk)
      aq[kk] = frag_ld(&qs[(mb + cl) * 64], kk * 32, lane);
#pragma unroll
    for (int j = 0; j < 5; ++j) {
      v8f a = vzero8();
#pragma unroll
      for (int kk = 0; kk < 2; ++kk) {
        v16bf bk = frag_ld(&kfull[(j * 16 + cl) * 64], kk * 32, lane);
        a = wmma_bf16(aq[kk], bk, a);
      }
#pragma unroll
      for (int e = 0; e < 8; ++e) a[e] *= 0.125f;     // 1/sqrt(64)
      sc[j] = a;
    }
  }
#pragma unroll
  for (int e = 0; e < 8; ++e) {
    float m = -1e30f;
#pragma unroll
    for (int j = 0; j < 5; ++j) m = fmaxf(m, sc[j][e]);
    m = fmaxf(m, __shfl_xor(m, 1, 32));
    m = fmaxf(m, __shfl_xor(m, 2, 32));
    m = fmaxf(m, __shfl_xor(m, 4, 32));
    m = fmaxf(m, __shfl_xor(m, 8, 32));
    float pv[5], s = 0.0f;
#pragma unroll
    for (int j = 0; j < 5; ++j) { pv[j] = __expf(sc[j][e] - m); s += pv[j]; }
    s += __shfl_xor(s, 1, 32);
    s += __shfl_xor(s, 2, 32);
    s += __shfl_xor(s, 4, 32);
    s += __shfl_xor(s, 8, 32);
    float inv = 1.0f / s;
    int row = mb + roff + e;
#pragma unroll
    for (int j = 0; j < 5; ++j)
      probs[row * KVPAD + j * 16 + cl] = (bf16_t)(pv[j] * inv);
  }
  __syncthreads();

  // ---- phase 4: out = probs @ v_full ----
  {
    v16bf ap[3];
#pragma unroll
    for (int kk = 0; kk < 3; ++kk)
      ap[kk] = frag_ld(&probs[(mb + cl) * KVPAD], kk * 32, lane);
#pragma unroll
    for (int j = 0; j < 4; ++j) {
      v8f a = vzero8();
#pragma unroll
      for (int kk = 0; kk < 3; ++kk) {
        v16bf bv = frag_ld(&vfullT[(j * 16 + cl) * KVPAD], kk * 32, lane);
        a = wmma_bf16(ap[kk], bv, a);
      }
#pragma unroll
      for (int e = 0; e < 8; ++e) {
        int rr = mb + roff + e;
        if (rr < NTOK)
          aout[((long)bw * NTOK + rr) * CDIM + h * DH + j * 16 + cl] = (bf16_t)a[e];
      }
    }
  }
}

// ---------------- GEMM 2: out = window_reverse(attn_out @ proj_w + b) ----------------
// Both tiles staged via async global->LDS: the k-loop is async-issue + WMMA only.

__global__ __launch_bounds__(256) void gemm_proj_kernel(
    const bf16_t* __restrict__ ain, const bf16_t* __restrict__ wpT,
    const float* __restrict__ bias, float* __restrict__ out) {
  __shared__ alignas(16) bf16_t As[2][128 * 40];
  __shared__ alignas(16) bf16_t Bs[2][128 * 40];
  const int tid = threadIdx.x, lane = tid & 31, wave = tid >> 5;
  const int Mbase = blockIdx.x * 128, Nbase = blockIdx.y * 128;
  const int wm = wave & 3, wn = wave >> 2;

  const int r = tid >> 2, c = (tid & 3) * 8;
  const bf16_t* __restrict__ srcA0 = ain + (long)(Mbase + r) * CDIM + c;
  const bf16_t* __restrict__ srcA1 = ain + (long)(Mbase + r + 64) * CDIM + c;
  const bf16_t* __restrict__ srcB0 = wpT + (long)(Nbase + r) * CDIM + c;
  const bf16_t* __restrict__ srcB1 = wpT + (long)(Nbase + r + 64) * CDIM + c;
  const int sA0 = r * 40 + c, sA1 = (r + 64) * 40 + c;
  const unsigned BUFSZ = 128 * 40 * 2;
  const unsigned ldsA0 = lds_off(&As[0][sA0]);
  const unsigned ldsA1 = lds_off(&As[0][sA1]);
  const unsigned ldsB0 = lds_off(&Bs[0][sA0]);
  const unsigned ldsB1 = lds_off(&Bs[0][sA1]);

  async_b128(ldsA0, srcA0);
  async_b128(ldsA1, srcA1);
  async_b128(ldsB0, srcB0);
  async_b128(ldsB1, srcB1);
  wait_async();
  __syncthreads();

  v8f acc[2][4];
#pragma unroll
  for (int i = 0; i < 2; ++i)
#pragma unroll
    for (int j = 0; j < 4; ++j) acc[i][j] = vzero8();

  const int arow0 = (wm * 32 + (lane & 15)) * 40;
  const int brow0 = (wn * 64 + (lane & 15)) * 40;

  int buf = 0;
  for (int kb0 = 0; kb0 < CDIM - 32; kb0 += 32) {
    const int knx = kb0 + 32;
    const unsigned nb = (unsigned)(buf ^ 1) * BUFSZ;
    async_b128(ldsA0 + nb, srcA0 + knx);
    async_b128(ldsA1 + nb, srcA1 + knx);
    async_b128(ldsB0 + nb, srcB0 + knx);
    async_b128(ldsB1 + nb, srcB1 + knx);

    const bf16_t* Ab = As[buf];
    const bf16_t* Bb = Bs[buf];
    v16bf af[2], bfr[4];
#pragma unroll
    for (int i = 0; i < 2; ++i) af[i] = frag_ld(Ab + arow0 + i * 16 * 40, 0, lane);
#pragma unroll
    for (int j = 0; j < 4; ++j) bfr[j] = frag_ld(Bb + brow0 + j * 16 * 40, 0, lane);
#pragma unroll
    for (int i = 0; i < 2; ++i)
#pragma unroll
      for (int j = 0; j < 4; ++j) acc[i][j] = wmma_bf16(af[i], bfr[j], acc[i][j]);

    wait_async();
    __syncthreads();
    buf ^= 1;
  }
  {
    const bf16_t* Ab = As[buf];
    const bf16_t* Bb = Bs[buf];
    v16bf af[2], bfr[4];
#pragma unroll
    for (int i = 0; i < 2; ++i) af[i] = frag_ld(Ab + arow0 + i * 16 * 40, 0, lane);
#pragma unroll
    for (int j = 0; j < 4; ++j) bfr[j] = frag_ld(Bb + brow0 + j * 16 * 40, 0, lane);
#pragma unroll
    for (int i = 0; i < 2; ++i)
#pragma unroll
      for (int j = 0; j < 4; ++j) acc[i][j] = wmma_bf16(af[i], bfr[j], acc[i][j]);
  }

  const int roff = (lane & 16) ? 8 : 0;
  const int cl = lane & 15;
#pragma unroll
  for (int i = 0; i < 2; ++i) {
    int row0 = Mbase + wm * 32 + i * 16 + roff;
    int bw = row0 / 49, t = row0 - bw * 49;
    int ty = t / 7, tx = t - ty * 7;
    int xrv[8];
#pragma unroll
    for (int e = 0; e < 8; ++e) {                    // incremental window reverse
      int b = bw >> 6, wi = bw & 63;
      xrv[e] = b * 3136 + ((wi >> 3) * 7 + ty) * 56 + (wi & 7) * 7 + tx;
      if (++tx == 7) { tx = 0; if (++ty == 7) { ty = 0; ++bw; } }
    }
#pragma unroll
    for (int j = 0; j < 4; ++j) {
      int cc = Nbase + wn * 64 + j * 16 + cl;
      float bv = bias[cc];
#pragma unroll
      for (int e = 0; e < 8; ++e)
        out[(long)xrv[e] * CDIM + cc] = acc[i][j][e] + bv;
    }
  }
}

// ---------------- host launch ----------------

extern "C" void kernel_launch(void* const* d_in, const int* in_sizes, int n_in,
                              void* d_out, int out_size, void* d_ws, size_t ws_size,
                              hipStream_t stream) {
  (void)in_sizes; (void)n_in; (void)out_size; (void)ws_size;
  const float* x       = (const float*)d_in[0];
  const float* qkv_w   = (const float*)d_in[1];
  const float* qkv_b   = (const float*)d_in[2];
  const float* E_k     = (const float*)d_in[3];
  const float* E_v     = (const float*)d_in[4];
  const float* gkw     = (const float*)d_in[5];
  const float* gvw     = (const float*)d_in[6];
  const float* proj_w  = (const float*)d_in[7];
  const float* proj_b  = (const float*)d_in[8];
  float* out = (float*)d_out;

  char* ws = (char*)d_ws;
  size_t off = 0;
  auto nxt = [&](size_t bytes) -> void* {
    void* p = ws + off;
    off += (bytes + 255) & ~(size_t)255;
    return p;
  };
  bf16_t* wqT  = (bf16_t*)nxt((size_t)CDIM * C3 * 2);
  bf16_t* wpT  = (bf16_t*)nxt((size_t)CDIM * CDIM * 2);
  bf16_t* ekt  = (bf16_t*)nxt((size_t)64 * 64 * 2);
  bf16_t* evt  = (bf16_t*)nxt((size_t)64 * 64 * 2);
  bf16_t* gkb  = (bf16_t*)nxt((size_t)NH * BANKN * DH * 2);
  bf16_t* gvb  = (bf16_t*)nxt((size_t)NH * BANKN * DH * 2);
  bf16_t* qbuf = (bf16_t*)nxt((size_t)BWH * NTOK * DH * 2);
  bf16_t* kTb  = (bf16_t*)nxt((size_t)BWH * DH * TPAD * 2);
  bf16_t* vTb  = (bf16_t*)nxt((size_t)BWH * DH * TPAD * 2);
  bf16_t* aout = (bf16_t*)nxt((size_t)MTOT * CDIM * 2);

  prep_kernel<<<2048, 256, 0, stream>>>(qkv_w, proj_w, E_k, E_v, gkw, gvw,
                                        wqT, wpT, ekt, evt, gkb, gvb);
  pad_kv_kernel<<<2048, 256, 0, stream>>>(kTb, vTb);
  dim3 g1(MTOT / 128, C3 / 128);
  gemm_qkv_kernel<<<g1, 256, 0, stream>>>(x, wqT, qkv_b, qbuf, kTb, vTb);
  attn_kernel<<<BWH, 128, 0, stream>>>(qbuf, kTb, vTb, ekt, evt, gkb, gvb, aout);
  dim3 g2(MTOT / 128, CDIM / 128);
  gemm_proj_kernel<<<g2, 256, 0, stream>>>(aout, wpT, proj_b, out);
}
